// LlamaAttention_67516885893724
// MI455X (gfx1250) — matrix operand order repaired and checked by reading, not compiled, not run on hardware
//
#include <hip/hip_runtime.h>
#include <math.h>
#include <stdint.h>

typedef __attribute__((ext_vector_type(16))) _Float16 v16h;
typedef __attribute__((ext_vector_type(8)))  float    v8f;
typedef unsigned int u32x4 __attribute__((ext_vector_type(4)));
typedef int          i32x4 __attribute__((ext_vector_type(4)));
typedef int          i32x8 __attribute__((ext_vector_type(8)));

#define T_SEQ   2048
#define HIDDEN_ 4096
#define NH      32
#define NKV     8
#define HD      128
#define QKV_LD  6144   // 4096 q + 1024 k + 1024 v

// ---- CDNA5 async global->LDS copy (ASYNCcnt path, ISA 15.18.3 op 98) -------
__device__ __forceinline__ void async_ld_b128(void* lds, const void* g)
{
  // VDST = LDS byte offset (aperture check uses addr[63:32]; low 32 bits of a
  // flat LDS pointer are the LDS offset, ISA 10.2).
  asm volatile("global_load_async_to_lds_b128 %0, %1, off"
               :: "v"((uint32_t)(uintptr_t)lds), "v"((uint64_t)(uintptr_t)g)
               : "memory");
}
__device__ __forceinline__ void wait_async0()
{
  asm volatile("s_wait_asynccnt 0x0" ::: "memory");
}
__device__ __forceinline__ void wait_tensor0()
{
#if __has_builtin(__builtin_amdgcn_s_wait_tensorcnt)
  __builtin_amdgcn_s_wait_tensorcnt(0);
#else
  asm volatile("s_wait_tensorcnt 0x0" ::: "memory");
#endif
}

// ---- Tensor Data Mover: 2D f16 tile global->LDS (ISA ch.8 D# layout) -------
// tile_w elems per row (row = 256B here), tile_h rows, row stride in elems.
// pad fields add 16B to the LDS address every 256B -> 272B LDS rows (=LDH).
__device__ __forceinline__ void tdm_load_tile_2d(
    uint32_t lds_off, const void* gptr,
    uint32_t tile_w, uint32_t tile_h, uint32_t stride_elems)
{
  uint64_t ga = (uint64_t)(uintptr_t)gptr;
  u32x4 g0;
  g0.x = 1u;                                  // count=1, user mode, no gather
  g0.y = lds_off;                             // lds_addr (bytes)
  g0.z = (uint32_t)ga;                        // global_addr[31:0]
  g0.w = (uint32_t)(ga >> 32) | (2u << 30);   // global_addr[56:32] | type=2
  i32x8 g1;
  g1[0] = (1 << 16)                           // data_size = 2 bytes
        | (1 << 20)                           // pad_enable
        | (5 << 22)                           // pad_interval: 2^5*8B = 256B
        | (3 << 25);                          // pad_amount: 4 DWORDs = 16B
  g1[1] = (int)(tile_w << 16);                // tensor_dim0[15:0]  @63:48
  g1[2] = (int)((tile_w >> 16) | (tile_h << 16));   // dim0 hi | dim1 lo
  g1[3] = (int)((tile_h >> 16) | (tile_w << 16));   // dim1 hi | tile_dim0
  g1[4] = (int)(tile_h & 0xFFFF);             // tile_dim1 | tile_dim2=0
  g1[5] = (int)stride_elems;                  // tensor_dim0_stride[31:0]
  g1[6] = 0;                                  // stride hi | dim1_stride lo
  g1[7] = 0;
  i32x4 z4 = {0, 0, 0, 0};
#if defined(__clang_major__) && __clang_major__ >= 23
  i32x8 z8 = {0, 0, 0, 0, 0, 0, 0, 0};
  __builtin_amdgcn_tensor_load_to_lds(g0, g1, z4, z4, z8, 0);
#else
  __builtin_amdgcn_tensor_load_to_lds(g0, g1, z4, z4, 0);
#endif
}

// ---------------------------------------------------------------------------
// GEMM: C[M,N] = A[M,K] @ B[N,K]^T  (f32 memory, f16 WMMA, f32 accumulate)
// Block tile 64(M) x 256(N), K-step 32, double-buffered LDS: global loads of
// tile k+1 overlap the 8 WMMAs of tile k; one barrier per K-step.
// ---------------------------------------------------------------------------
__global__ __launch_bounds__(256)
void gemm_nt_wmma(const float* __restrict__ A, const float* __restrict__ B,
                  float* __restrict__ C, int M, int N, int K)
{
  constexpr int BM = 64, BN = 256, BK = 32, LD = BK + 8;
  __shared__ _Float16 sA[2][BM * LD];
  __shared__ _Float16 sB[2][BN * LD];

  const int tid   = threadIdx.x;
  const int wave  = tid >> 5;
  const int lane  = tid & 31;
  const int lhalf = lane & 15;
  const int hsel  = lane >> 4;        // 0: lanes 0-15, 1: lanes 16-31
  const int wr = wave >> 2;           // 0..1
  const int wc = wave & 3;            // 0..3
  const int m0 = blockIdx.y * BM + wr * 32;
  const int n0 = blockIdx.x * BN + wc * 64;

  const int arow = tid >> 2, acol = (tid & 3) * 8;   // A: 8 floats/thread
  const float* aptr = A + (size_t)(blockIdx.y * BM + arow) * K + acol;
  const float* bptr = B + (size_t)(blockIdx.x * BN + tid) * K; // B: 32/thread

  float ra[8], rb[32];
  auto load_tile = [&](int k0) {
    #pragma unroll
    for (int i = 0; i < 8; ++i)  ra[i] = aptr[k0 + i];
    #pragma unroll
    for (int i = 0; i < 32; ++i) rb[i] = bptr[k0 + i];
  };
  auto store_tile = [&](int buf) {
    _Float16* da = &sA[buf][arow * LD + acol];
    #pragma unroll
    for (int i = 0; i < 8; ++i)  da[i] = (_Float16)ra[i];
    _Float16* db = &sB[buf][tid * LD];
    #pragma unroll
    for (int i = 0; i < 32; ++i) db[i] = (_Float16)rb[i];
  };

  load_tile(0);
  store_tile(0);
  __syncthreads();

  v8f acc[2][4] = {};
  const int nk = K / BK;
  for (int ki = 0; ki < nk; ++ki) {
    const int cur = ki & 1;
    if (ki + 1 < nk) load_tile((ki + 1) * BK);   // prefetch next tile

    // A fragment (16-bit 16x32): lane = row; lanes 0-15: K 0..7/16..23,
    // lanes 16-31: K 8..15/24..31.  B fragment (32x16): lane = column.
    v16h af[2], bf[4];
    #pragma unroll
    for (int t = 0; t < 2; ++t) {
      const _Float16* p = &sA[cur][(wr * 32 + t * 16 + lhalf) * LD + hsel * 8];
      #pragma unroll
      for (int i = 0; i < 8; ++i) { af[t][i] = p[i]; af[t][8 + i] = p[16 + i]; }
    }
    #pragma unroll
    for (int t = 0; t < 4; ++t) {
      const _Float16* p = &sB[cur][(wc * 64 + t * 16 + lhalf) * LD + hsel * 8];
      #pragma unroll
      for (int i = 0; i < 8; ++i) { bf[t][i] = p[i]; bf[t][8 + i] = p[16 + i]; }
    }
    #pragma unroll
    for (int tm = 0; tm < 2; ++tm)
      #pragma unroll
      for (int tn = 0; tn < 4; ++tn)
        acc[tm][tn] = __builtin_amdgcn_wmma_f32_16x16x32_f16(
            false, af[tm], false, bf[tn], (short)0, acc[tm][tn], false, false);

    if (ki + 1 < nk) store_tile(cur ^ 1);        // fill other buffer
    __syncthreads();
  }

  // C/D layout: VGPR r holds row (r + 8*hsel), lane half gives column 0..15
  #pragma unroll
  for (int tm = 0; tm < 2; ++tm)
    #pragma unroll
    for (int tn = 0; tn < 4; ++tn) {
      int n = n0 + tn * 16 + lhalf;
      #pragma unroll
      for (int r = 0; r < 8; ++r) {
        int m = m0 + tm * 16 + r + hsel * 8;
        C[(size_t)m * N + n] = acc[tm][tn][r];
      }
    }
}

// ---------------------------------------------------------------------------
// RoPE on q (heads 0..31) and k (heads 32..39); plain f16 convert for v
// (heads 40..47). Writes the rotated f16 mirror consumed by attention.
// ---------------------------------------------------------------------------
__global__ __launch_bounds__(256)
void rope_to_f16(const float* __restrict__ qkv, _Float16* __restrict__ qkv16,
                 const int* __restrict__ pos)
{
  int idx = blockIdx.x * blockDim.x + threadIdx.x;   // T * 48 * 64 threads
  int d = idx & 63;
  int h = (idx >> 6) % 48;
  int t = idx / (64 * 48);
  if (t >= T_SEQ) return;
  const float* base = qkv + (size_t)t * QKV_LD + h * HD;
  _Float16* b16 = qkv16 + (size_t)t * QKV_LD + h * HD;
  float x1 = base[d], x2 = base[d + 64];
  if (h < NH + NKV) {
    float inv = powf(500000.0f, -(float)d * (1.0f / 64.0f));
    float ang = (float)pos[t] * inv;
    float s, c;
    sincosf(ang, &s, &c);
    b16[d]      = (_Float16)(x1 * c - x2 * s);
    b16[d + 64] = (_Float16)(x2 * c + x1 * s);
  } else {
    b16[d]      = (_Float16)x1;
    b16[d + 64] = (_Float16)x2;
  }
}

// ---------------------------------------------------------------------------
// Flash attention. Grid (32 heads, 16 q-blocks of 128). 8 waves x 16 queries.
// K chunks staged by the Tensor Data Mover (one tensor_load_to_lds per chunk,
// TENSORcnt); V chunks staged by per-lane GLOBAL_LOAD_ASYNC_TO_LDS_B128
// (ASYNCcnt). Double-buffered: chunk kb+1 DMA overlaps chunk kb's 16 WMMAs.
// ---------------------------------------------------------------------------
__global__ __launch_bounds__(256)
void attn_wmma(const _Float16* __restrict__ qkv16, float* __restrict__ Oout)
{
  constexpr int LDH = HD + 8;                 // 136 halves = 272B rows
  __shared__ _Float16 sK[2][32 * LDH];
  __shared__ _Float16 sV[2][32 * LDH];
  __shared__ _Float16 sP[8][16 * 40];         // per-wave P transpose buffer

  const int h    = blockIdx.x;
  const int qb   = blockIdx.y;
  const int kvh  = h >> 2;                    // GROUP = 4
  const int tid  = threadIdx.x;
  const int wave = tid >> 5;
  const int lane = tid & 31;
  const int lhalf = lane & 15;
  const int hsel  = lane >> 4;
  const int q0 = qb * 128 + wave * 16;

  const float scale = 0.088388347648318447f;  // 1/sqrt(128)

  auto stage = [&](int kb32, int buf) {
    const int kstart = kb32 * 32;
    if (wave == 0) {     // K chunk 32x128 f16 via TDM (EXEC-independent)
      const _Float16* ks = qkv16 + (size_t)kstart * QKV_LD
                                 + (size_t)NH * HD + kvh * HD;
      tdm_load_tile_2d((uint32_t)(uintptr_t)&sK[buf][0], ks, HD, 32, QKV_LD);
    }
    { // V chunk 32x128 f16 via per-lane async b128 (2 ops/thread)
      int r = tid >> 3;
      int c = (tid & 7) * 16;
      const _Float16* vs = qkv16 + (size_t)(kstart + r) * QKV_LD
                                 + (size_t)(NH + NKV) * HD + kvh * HD + c;
      _Float16* vd = &sV[buf][r * LDH + c];
      async_ld_b128(vd,     vs);
      async_ld_b128(vd + 8, vs + 8);
    }
  };

  // Q fragments for all 4 K-steps of head_dim (pre-scaled, f16 source)
  v16h qf[4];
  {
    const _Float16* qrow = qkv16 + (size_t)(q0 + lhalf) * QKV_LD + h * HD;
    const int kb = hsel * 8;
    #pragma unroll
    for (int s2 = 0; s2 < 4; ++s2)
      #pragma unroll
      for (int i = 0; i < 8; ++i) {
        qf[s2][i]     = (_Float16)((float)qrow[s2 * 32 + kb + i] * scale);
        qf[s2][8 + i] = (_Float16)((float)qrow[s2 * 32 + kb + 16 + i] * scale);
      }
  }

  v8f oacc[8] = {};                // 16 x 128 output accumulator
  float mrow[8], lrow[8];
  #pragma unroll
  for (int r = 0; r < 8; ++r) { mrow[r] = -1e30f; lrow[r] = 0.0f; }

  const int nkb = (qb + 1) * 4;    // key blocks of 32 (uniform per workgroup)
  stage(0, 0);
  for (int kb32 = 0; kb32 < nkb; ++kb32) {
    const int kstart = kb32 * 32;
    const int cur = kb32 & 1;
    if (wave == 0) wait_tensor0();   // this wave's TDM for chunk kb32
    wait_async0();                   // this wave's V copies for chunk kb32
    __syncthreads();                 // all staging done; prev compute done
    if (kb32 + 1 < nkb) stage(kb32 + 1, cur ^ 1);  // overlap next DMA

    // S = Q @ K^T : two 16x16 key tiles, 4 K-steps over head_dim
    v8f sacc[2] = {};
    #pragma unroll
    for (int t = 0; t < 2; ++t)
      #pragma unroll
      for (int s2 = 0; s2 < 4; ++s2) {
        v16h bf;
        const _Float16* p = &sK[cur][(t * 16 + lhalf) * LDH + hsel * 8 + s2 * 32];
        #pragma unroll
        for (int i = 0; i < 8; ++i) { bf[i] = p[i]; bf[8 + i] = p[16 + i]; }
        sacc[t] = __builtin_amdgcn_wmma_f32_16x16x32_f16(
            false, qf[s2], false, bf, (short)0, sacc[t], false, false);
      }

    // causal mask (element (r,lane): q = q0+r+8*hsel, key = kstart+16t+lhalf)
    #pragma unroll
    for (int t = 0; t < 2; ++t) {
      int key = kstart + t * 16 + lhalf;
      #pragma unroll
      for (int r = 0; r < 8; ++r) {
        int q = q0 + r + hsel * 8;
        if (key > q) sacc[t][r] = -1e30f;
      }
    }

    // online softmax: row reductions across the 16 lanes of each half
    #pragma unroll
    for (int r = 0; r < 8; ++r) {
      float v = fmaxf(sacc[0][r], sacc[1][r]);
      #pragma unroll
      for (int off = 1; off < 16; off <<= 1) v = fmaxf(v, __shfl_xor(v, off, 32));
      float nm = fmaxf(mrow[r], v);
      float p0 = __expf(sacc[0][r] - nm);
      float p1 = __expf(sacc[1][r] - nm);
      sacc[0][r] = p0; sacc[1][r] = p1;
      float rs = p0 + p1;
      #pragma unroll
      for (int off = 1; off < 16; off <<= 1) rs += __shfl_xor(rs, off, 32);
      float alpha = __expf(mrow[r] - nm);
      lrow[r] = lrow[r] * alpha + rs;
      mrow[r] = nm;
      #pragma unroll
      for (int j = 0; j < 8; ++j) oacc[j][r] *= alpha;
    }

    // P (C/D layout) -> A-fragment layout via per-wave LDS round-trip
    _Float16* pb = sP[wave];
    #pragma unroll
    for (int t = 0; t < 2; ++t)
      #pragma unroll
      for (int r = 0; r < 8; ++r)
        pb[(r + hsel * 8) * 40 + t * 16 + lhalf] = (_Float16)sacc[t][r];
    asm volatile("s_wait_dscnt 0x0" ::: "memory");   // same-wave DS RAW fence

    v16h pf;
    {
      const _Float16* p = &pb[lhalf * 40 + hsel * 8];
      #pragma unroll
      for (int i = 0; i < 8; ++i) { pf[i] = p[i]; pf[8 + i] = p[16 + i]; }
    }
    // O += P @ V : 8 head-dim tiles, K = 32 keys
    #pragma unroll
    for (int j = 0; j < 8; ++j) {
      v16h vf;
      #pragma unroll
      for (int i = 0; i < 16; ++i)
        vf[i] = sV[cur][((i & 8) * 2 + hsel * 8 + (i & 7)) * LDH + j * 16 + lhalf];
      oacc[j] = __builtin_amdgcn_wmma_f32_16x16x32_f16(
          false, pf, false, vf, (short)0, oacc[j], false, false);
    }
  }

  // finalize: divide by row sums and store (o is [T, 4096], col = h*128 + d)
  #pragma unroll
  for (int j = 0; j < 8; ++j) {
    int col = h * HD + j * 16 + lhalf;
    #pragma unroll
    for (int r = 0; r < 8; ++r) {
      int q = q0 + r + hsel * 8;
      Oout[(size_t)q * (NH * HD) + col] = oacc[j][r] / lrow[r];
    }
  }
}

// ---------------------------------------------------------------------------
extern "C" void kernel_launch(void* const* d_in, const int* in_sizes, int n_in,
                              void* d_out, int out_size, void* d_ws, size_t ws_size,
                              hipStream_t stream)
{
  (void)in_sizes; (void)n_in; (void)out_size; (void)ws_size;
  const int*   positions = (const int*)d_in[0];
  const float* hidden    = (const float*)d_in[1];
  const float* w_qkv     = (const float*)d_in[2];
  const float* w_o       = (const float*)d_in[3];
  float* out = (float*)d_out;

  float*    qkv   = (float*)d_ws;                         // 2048*6144 f32
  float*    oat   = qkv + (size_t)T_SEQ * QKV_LD;         // 2048*4096 f32
  _Float16* qkv16 = (_Float16*)(oat + (size_t)T_SEQ * HIDDEN_); // 2048*6144 f16

  dim3 blk(256);
  // 1) qkv = hidden @ w_qkv^T
  gemm_nt_wmma<<<dim3(QKV_LD / 256, T_SEQ / 64), blk, 0, stream>>>(
      hidden, w_qkv, qkv, T_SEQ, QKV_LD, HIDDEN_);
  // 2) RoPE + f16 mirror (q,k rotated; v converted)
  rope_to_f16<<<dim3((T_SEQ * 48 * 64) / 256), blk, 0, stream>>>(
      qkv, qkv16, positions);
  // 3) flash attention per (head, 128-query block)
  attn_wmma<<<dim3(NH, T_SEQ / 128), blk, 0, stream>>>(qkv16, oat);
  // 4) out = o @ w_o^T
  gemm_nt_wmma<<<dim3(HIDDEN_ / 256, T_SEQ / 64), blk, 0, stream>>>(
      oat, w_o, out, T_SEQ, HIDDEN_, HIDDEN_);
}
